// WindowAttention3D_74663711474174
// MI455X (gfx1250) — compile-verified, matrix operand order checked by reading, not run on hardware
//
#include <hip/hip_runtime.h>
#include <hip/hip_bf16.h>

typedef _Float16 h16v __attribute__((ext_vector_type(16)));
typedef float    f8v  __attribute__((ext_vector_type(8)));

#define DIM   128
#define HEADS 4
#define HD    32
#define NTOK  98
#define MPAD  112
#define NWIN  64

// LDS byte offsets (one dynamic block, regions aliased across phases)
#define OFF_A     0          // region A: xh (phase1) / sc (phase2)
#define OFF_QH    51968
#define OFF_KH    82432
#define OFF_VT    112896
#define OFF_PH    147712
#define OFF_OH    178176
#define LDS_TOTAL 208640

// Workspace layout (f16): qkvT[384][128] then projT[128][128]
#define WS_QKVT_ELEMS (384 * 128)
#define WS_PROJT_ELEMS (128 * 128)

// Load a 16x32 f16 WMMA fragment from a row-major matrix (LDS or global).
// Works for A (row-major A) and for B when the matrix stored is B^T row-major.
// Layout per CDNA5 ISA: lanes 0-15 hold row M=lane&15, K in
// {k0..k0+7, k0+16..k0+23} with k0 = 0 for lanes<16 and 8 for lanes>=16.
__device__ __forceinline__ h16v frag_ld(const _Float16* M, int stride,
                                        int rowBase, int kBase, int lane) {
  const int row = rowBase + (lane & 15);
  const int k0  = kBase + ((lane >> 4) << 3);
  const _Float16* p = M + row * stride + k0;
  h16v r;
#pragma unroll
  for (int i = 0; i < 8; ++i) { r[i] = p[i]; r[i + 8] = p[i + 16]; }
  return r;
}

// One-shot weight pre-pass: transpose + f32->f16 so the main kernel can load
// B fragments straight from global (stays L2-resident: 128 KB total).
__global__ __launch_bounds__(256)
void prep_weights_kernel(const float* __restrict__ qkv_w,
                         const float* __restrict__ proj_w,
                         _Float16* __restrict__ ws) {
  int e = blockIdx.x * 256 + threadIdx.x;
  if (e < WS_QKVT_ELEMS) {
    int n = e >> 7, k = e & 127;                 // qkvT[n][k] = qkv_w[k][n]
    ws[e] = (_Float16)qkv_w[k * 384 + n];
  } else if (e < WS_QKVT_ELEMS + WS_PROJT_ELEMS) {
    int e2 = e - WS_QKVT_ELEMS;
    int n = e2 >> 7, k = e2 & 127;               // projT[n][k] = proj_w[k][n]
    ws[e] = (_Float16)proj_w[k * 128 + n];
  }
}

__global__ __launch_bounds__(256, 1)
void win_attn3d_kernel(const float* __restrict__ x, const float* __restrict__ mask,
                       const _Float16* __restrict__ qkvT, const float* __restrict__ qkv_b,
                       const _Float16* __restrict__ projT, const float* __restrict__ proj_b,
                       const float* __restrict__ bias_table, const int* __restrict__ rel_index,
                       float* __restrict__ out) {
  extern __shared__ char smem[];
  _Float16* xh = (_Float16*)(smem + OFF_A);   // 112 x 128, stride 136 (f16)
  float*    sc = (float*)(smem + OFF_A);      // 112 x 112, stride 116 (f32)
  _Float16* qh = (_Float16*)(smem + OFF_QH);  // 112 x 128, stride 136
  _Float16* kh = (_Float16*)(smem + OFF_KH);  // 112 x 128, stride 136
  _Float16* vT = (_Float16*)(smem + OFF_VT);  // 128 x 128 (d-major), stride 136
  _Float16* ph = (_Float16*)(smem + OFF_PH);  // 112 x 128, stride 136 (probs)
  _Float16* oh = (_Float16*)(smem + OFF_OH);  // 112 x 128, stride 136 (attn out)

  const int blk  = blockIdx.x;
  const int w    = blk & (NWIN - 1);
  const int tid  = threadIdx.x;
  const int wv   = tid >> 5;                  // wave id == column tile in GEMMs
  const int lane = tid & 31;
  const float* mk = mask + (size_t)w * NTOK * NTOK;

  // stage x -> xh (f16), rows 98..111 zero
  for (int e = tid; e < MPAD * DIM; e += 256) {
    int i = e >> 7, c = e & 127;
    float v = (i < NTOK) ? x[((size_t)blk * NTOK + i) * DIM + c] : 0.f;
    xh[i * 136 + c] = (_Float16)v;
  }
  // zero vT token-padding columns t in [112,128)
  for (int e = tid; e < 128 * 16; e += 256) {
    int d = e >> 4, t = 112 + (e & 15);
    vT[d * 136 + t] = (_Float16)0.f;
  }
  __syncthreads();

  // ---------------- Phase 1: QKV = x @ qkv_w + qkv_b ----------------
  // B fragments straight from pre-transposed f16 weights in global (L2-hot).
  for (int g = 0; g < 3; ++g) {               // 0:q 1:k 2:v (col groups of 128)
    const _Float16* wg = qkvT + (size_t)(g * 128) * 128;  // rows = out col, stride 128
    f8v acc[7];
#pragma unroll
    for (int t = 0; t < 7; ++t)
#pragma unroll
      for (int r = 0; r < 8; ++r) acc[t][r] = 0.f;

#pragma unroll
    for (int ks = 0; ks < 4; ++ks) {
      h16v b = frag_ld(wg, 128, wv * 16, ks * 32, lane);  // one B frag per wave
      if (ks < 3)
        __builtin_prefetch((const void*)(wg + (wv * 16 + (lane & 15)) * 128 +
                                         (ks + 1) * 32), 0, 1);
#pragma unroll
      for (int t = 0; t < 7; ++t) {                       // 7 row tiles
        h16v a = frag_ld(xh, 136, t * 16, ks * 32, lane);
        acc[t] = __builtin_amdgcn_wmma_f32_16x16x32_f16(false, a, false, b,
                                                        (short)0, acc[t], false, false);
      }
    }
    // epilogue: +bias, q pre-scaled, v stored transposed (ct == wv)
    const int nl = lane & 15, mh = (lane >> 4) << 3;
    const int colG = wv * 16 + nl;
    const float bias = qkv_b[g * 128 + colG];
#pragma unroll
    for (int t = 0; t < 7; ++t) {
#pragma unroll
      for (int r = 0; r < 8; ++r) {
        int rowG = t * 16 + mh + r;
        float v = acc[t][r] + bias;
        if (g == 0)      qh[rowG * 136 + colG] = (_Float16)(v * 0.1767766953f); // 32^-0.5
        else if (g == 1) kh[rowG * 136 + colG] = (_Float16)v;
        else             vT[colG * 136 + rowG] = (_Float16)v;
      }
    }
  }
  __syncthreads();

  // ---------------- Phase 2: per-head attention ----------------
  for (int h = 0; h < HEADS; ++h) {
    // scores S = q_scaled @ k^T + rel_bias + mask  (single k-step, hd=32)
#pragma unroll
    for (int t = 0; t < 7; ++t) {
      int tile = wv + t * 8;            // 49 tiles: 7 x 7
      if (tile < 49) {
        int rt = tile / 7, ct = tile % 7;
        h16v a = frag_ld(qh, 136, rt * 16, h * HD, lane);
        h16v b = frag_ld(kh, 136, ct * 16, h * HD, lane);
        f8v c;
#pragma unroll
        for (int r = 0; r < 8; ++r) c[r] = 0.f;
        c = __builtin_amdgcn_wmma_f32_16x16x32_f16(false, a, false, b,
                                                   (short)0, c, false, false);
        int nl = lane & 15, mh = (lane >> 4) << 3;
        int j = ct * 16 + nl;
#pragma unroll
        for (int r = 0; r < 8; ++r) {
          int i = rt * 16 + mh + r;
          float s = c[r];
          if (i < NTOK && j < NTOK) {
            int bi = rel_index[i * NTOK + j];
            s += bias_table[bi * HEADS + h] + mk[i * NTOK + j];
          }
          sc[i * 116 + j] = s;
        }
      }
    }
    __syncthreads();

    // softmax: one wave per row, lanes parallel over columns
    for (int rr = 0; rr < 14; ++rr) {
      int i = wv + rr * 8;
      float vals[4], m = -1e30f;
#pragma unroll
      for (int it = 0; it < 4; ++it) {
        int cix = lane + it * 32;
        vals[it] = (cix < NTOK) ? sc[i * 116 + cix] : -1e30f;
        m = fmaxf(m, vals[it]);
      }
#pragma unroll
      for (int off = 16; off > 0; off >>= 1) m = fmaxf(m, __shfl_xor(m, off, 32));
      float sum = 0.f;
#pragma unroll
      for (int it = 0; it < 4; ++it) { vals[it] = __expf(vals[it] - m); sum += vals[it]; }
#pragma unroll
      for (int off = 16; off > 0; off >>= 1) sum += __shfl_xor(sum, off, 32);
      float inv = 1.f / sum;
#pragma unroll
      for (int it = 0; it < 4; ++it) {
        int cix = lane + it * 32;
        ph[i * 136 + cix] = (cix < NTOK) ? (_Float16)(vals[it] * inv) : (_Float16)0.f;
      }
    }
    __syncthreads();

    // out_h = P @ V  (K = 128 padded tokens, zeros beyond 98)
#pragma unroll
    for (int t = 0; t < 2; ++t) {
      int tile = wv + t * 8;            // 14 tiles: 7 row x 2 col
      if (tile < 14) {
        int rt = tile >> 1, cn = tile & 1;
        f8v c;
#pragma unroll
        for (int r = 0; r < 8; ++r) c[r] = 0.f;
#pragma unroll
        for (int ks = 0; ks < 4; ++ks) {
          h16v a = frag_ld(ph, 136, rt * 16, ks * 32, lane);
          h16v b = frag_ld(vT, 136, h * HD + cn * 16, ks * 32, lane);
          c = __builtin_amdgcn_wmma_f32_16x16x32_f16(false, a, false, b,
                                                     (short)0, c, false, false);
        }
        int nl = lane & 15, mh = (lane >> 4) << 3;
#pragma unroll
        for (int r = 0; r < 8; ++r) {
          int i = rt * 16 + mh + r;
          oh[i * 136 + h * HD + cn * 16 + nl] = (_Float16)c[r];
        }
      }
    }
    __syncthreads();
  }

  // ---------------- Phase 3: out = oh @ proj_w + proj_b ----------------
  {
    f8v acc[7];
#pragma unroll
    for (int t = 0; t < 7; ++t)
#pragma unroll
      for (int r = 0; r < 8; ++r) acc[t][r] = 0.f;

#pragma unroll
    for (int ks = 0; ks < 4; ++ks) {
      h16v b = frag_ld(projT, 128, wv * 16, ks * 32, lane);
      if (ks < 3)
        __builtin_prefetch((const void*)(projT + (wv * 16 + (lane & 15)) * 128 +
                                         (ks + 1) * 32), 0, 1);
#pragma unroll
      for (int t = 0; t < 7; ++t) {
        h16v a = frag_ld(oh, 136, t * 16, ks * 32, lane);
        acc[t] = __builtin_amdgcn_wmma_f32_16x16x32_f16(false, a, false, b,
                                                        (short)0, acc[t], false, false);
      }
    }
    const int nl = lane & 15, mh = (lane >> 4) << 3;
    const int col = wv * 16 + nl;
    const float pb = proj_b[col];
#pragma unroll
    for (int t = 0; t < 7; ++t) {
#pragma unroll
      for (int r = 0; r < 8; ++r) {
        int i = t * 16 + mh + r;
        if (i < NTOK) out[((size_t)blk * NTOK + i) * DIM + col] = acc[t][r] + pb;
      }
    }
  }
}

extern "C" void kernel_launch(void* const* d_in, const int* in_sizes, int n_in,
                              void* d_out, int out_size, void* d_ws, size_t ws_size,
                              hipStream_t stream) {
  const float* x          = (const float*)d_in[0];
  const float* mask       = (const float*)d_in[1];
  const float* qkv_w      = (const float*)d_in[2];
  const float* qkv_b      = (const float*)d_in[3];
  const float* proj_w     = (const float*)d_in[4];
  const float* proj_b     = (const float*)d_in[5];
  const float* bias_table = (const float*)d_in[6];
  const int*   rel_index  = (const int*)d_in[7];
  float* out = (float*)d_out;

  _Float16* wsT   = (_Float16*)d_ws;               // 128 KB of f16 weights
  const _Float16* qkvT = wsT;
  const _Float16* projT = wsT + WS_QKVT_ELEMS;

  const int nblk = in_sizes[0] / (NTOK * DIM);     // 4096 windows

  // Opt in to >64KB dynamic LDS (no-op for compile-only; not a stream op,
  // safe under graph capture).
  (void)hipFuncSetAttribute((const void*)win_attn3d_kernel,
                            hipFuncAttributeMaxDynamicSharedMemorySize, LDS_TOTAL);

  // Pass 1: transpose + convert weights to f16 once (L2-resident afterwards).
  prep_weights_kernel<<<(WS_QKVT_ELEMS + WS_PROJT_ELEMS + 255) / 256, 256, 0, stream>>>(
      qkv_w, proj_w, wsT);

  // Pass 2: fused window attention.
  win_attn3d_kernel<<<nblk, 256, LDS_TOTAL, stream>>>(
      x, mask, qkvT, qkv_b, projT, proj_b, bias_table, rel_index, out);
}
